// _self_attention_layer_46849503264775
// MI455X (gfx1250) — matrix operand order repaired and checked by reading, not compiled, not run on hardware
//
#include <hip/hip_runtime.h>
#include <stdint.h>

typedef __attribute__((ext_vector_type(16))) _Float16 v16h;
typedef __attribute__((ext_vector_type(8)))  _Float16 v8h;
typedef __attribute__((ext_vector_type(4)))  _Float16 v4h;
typedef __attribute__((ext_vector_type(8)))  float    v8f;

#define B_  8
#define S_  2048
#define D_  1024
#define H_  128
#define SCALE_ 0.03125f        // 1024^-0.5 folded into Q
#define NEG_INF_ (-1.0e9f)

#define WT_E  (3 * H_ * D_)    // f16 elems: transposed weights [m][h][d]
#define QK_E  (B_ * S_ * H_)   // f16 elems per Q / K / Vt buffer

// -------- gfx1250 async global->LDS copy (ASYNCcnt path) -------------------
#if defined(__gfx1250__) && __has_builtin(__builtin_amdgcn_global_load_async_to_lds_b128)
#define USE_ASYNC_LDS 1
typedef int v4i __attribute__((__vector_size__(4 * sizeof(int))));
typedef __attribute__((address_space(1))) v4i v4i_g;   // global b128 chunk
typedef __attribute__((address_space(3))) v4i v4i_l;   // LDS b128 chunk
__device__ __forceinline__ void async_cp16(const void* g, void* l) {
  // flat LDS pointer low 32 bits are the LDS byte address (ISA 10.2 aperture)
  __builtin_amdgcn_global_load_async_to_lds_b128(
      (v4i_g*)(uintptr_t)g, (v4i_l*)(uint32_t)(uintptr_t)l, 0, 0);
}
#endif

// ---------------------------------------------------------------------------
// Kernel 0: W[D,H] fp32 -> Wt[m][H][D] f16 (transpose + convert, tiny)
// ---------------------------------------------------------------------------
__global__ void wt_transpose_kernel(const float* __restrict__ Wq,
                                    const float* __restrict__ Wk,
                                    const float* __restrict__ Wv,
                                    _Float16* __restrict__ Wt) {
  int idx = blockIdx.x * blockDim.x + threadIdx.x;
  if (idx >= 3 * D_ * H_) return;
  int m = idx / (D_ * H_);
  int r = idx % (D_ * H_);
  int d = r / H_;
  int h = r % H_;
  const float* W = (m == 0) ? Wq : (m == 1) ? Wk : Wv;
  Wt[(size_t)m * H_ * D_ + (size_t)h * D_ + d] = (_Float16)W[(size_t)d * H_ + h];
}

// ---------------------------------------------------------------------------
// Kernel 1: fused QKV projection.  One block = 16 rows of x; 8 waves each own
// a 16-wide column tile of Q, K and V.  x tile staged to LDS as f16.
// Q scaled by D^-0.5 row-major, K row-major, V transposed to [b][h][s].
// ---------------------------------------------------------------------------
__global__ void __launch_bounds__(256) qkv_proj_kernel(
    const float* __restrict__ x,
    const _Float16* __restrict__ Wt,
    _Float16* __restrict__ Qb,
    _Float16* __restrict__ Kb,
    _Float16* __restrict__ Vt) {
  __shared__ __align__(16) _Float16 xs[16][1032];   // 1032 = 1024 + 8 pad
  const int t = threadIdx.x;
  const int rowbase = blockIdx.x * 16;

  // stage 16x1024 fp32 -> f16 into LDS (float4 loads, ds_store_b64)
  const float* xb = x + (size_t)rowbase * D_;
  #pragma unroll
  for (int it = 0; it < 16; ++it) {
    int vi = t + it * 256;          // float4 index in [0,4096)
    int r  = vi >> 8;
    int c4 = vi & 255;
    float4 f = ((const float4*)(xb + (size_t)r * D_))[c4];
    v4h h4;
    h4[0] = (_Float16)f.x; h4[1] = (_Float16)f.y;
    h4[2] = (_Float16)f.z; h4[3] = (_Float16)f.w;
    *(v4h*)&xs[r][c4 * 4] = h4;
  }
  __syncthreads();

  const int wave = t >> 5;
  const int lane = t & 31;
  const int n  = lane & 15;
  const int hh = lane >> 4;
  const int hbase = wave * 16;

  v8f acc0 = {}; v8f acc1 = {}; v8f acc2 = {};
  const _Float16* W0 = Wt + (size_t)(hbase + n) * D_;
  const _Float16* W1 = W0 + (size_t)H_ * D_;
  const _Float16* W2 = W1 + (size_t)H_ * D_;

  #pragma unroll 4
  for (int k0 = 0; k0 < D_; k0 += 32) {
    v8h lo = *(const v8h*)&xs[n][k0 + hh * 16];
    v8h hi = *(const v8h*)&xs[n][k0 + hh * 16 + 8];
    v16h a;
    #pragma unroll
    for (int j = 0; j < 8; ++j) { a[j] = lo[j]; a[j + 8] = hi[j]; }
    v16h b0 = *(const v16h*)(W0 + k0 + hh * 16);
    v16h b1 = *(const v16h*)(W1 + k0 + hh * 16);
    v16h b2 = *(const v16h*)(W2 + k0 + hh * 16);
    acc0 = __builtin_amdgcn_wmma_f32_16x16x32_f16(false, a, false, b0, (short)0, acc0, false, false);
    acc1 = __builtin_amdgcn_wmma_f32_16x16x32_f16(false, a, false, b1, (short)0, acc1, false, false);
    acc2 = __builtin_amdgcn_wmma_f32_16x16x32_f16(false, a, false, b2, (short)0, acc2, false, false);
  }

  #pragma unroll
  for (int i = 0; i < 8; ++i) {
    int row = rowbase + i + 8 * hh;          // global row in [B*S)
    int b   = row / S_;
    int s   = row % S_;
    Qb[(size_t)row * H_ + hbase + n] = (_Float16)(acc0[i] * SCALE_);
    Kb[(size_t)row * H_ + hbase + n] = (_Float16)acc1[i];
    Vt[(size_t)b * H_ * S_ + (size_t)(hbase + n) * S_ + s] = (_Float16)acc2[i];
  }
}

// ---------------------------------------------------------------------------
// Kernel 2: causal flash attention.  Block = 4 waves = 64 queries (one batch).
// Per 32-key step: the whole block stages the K tile (32x128 f16) and V tile
// (128x32 f16, from transposed V) into LDS with async global->LDS copies
// (ASYNCcnt), then each wave runs 8 WMMAs (scores), online softmax, P C->A
// relayout through LDS, and 8 WMMAs (PV).  Loop bound is block-uniform so
// __syncthreads() is legal; out-of-range tiles are fully masked.
// ---------------------------------------------------------------------------
__global__ void __launch_bounds__(128) flash_attn_kernel(
    const _Float16* __restrict__ Qb,
    const _Float16* __restrict__ Kb,
    const _Float16* __restrict__ Vt,
    float* __restrict__ out) {
  __shared__ __align__(16) _Float16 Klds[32][136];   // keys x H (+pad)
  __shared__ __align__(16) _Float16 Vlds[128][40];   // H x keys (+pad)
  __shared__ __align__(16) _Float16 Pst[4][16][40];  // per-wave P 16x32 (+pad)

  const int t    = threadIdx.x;
  const int wave = t >> 5;
  const int lane = t & 31;
  const int n  = lane & 15;
  const int hh = lane >> 4;

  const int qtile = blockIdx.x * 4 + wave;
  const int growq = qtile * 16;            // global query row base
  const int b     = (blockIdx.x * 64) / S_;
  const int qloc  = growq % S_;            // query s within batch

  // resident Q tile in A-operand layout: 4 chunks of K=32 across H=128
  v16h qa[4];
  const _Float16* qrow = Qb + (size_t)(growq + n) * H_;
  #pragma unroll
  for (int c = 0; c < 4; ++c) {
    v8h lo = *(const v8h*)(qrow + c * 32 + hh * 16);
    v8h hi = *(const v8h*)(qrow + c * 32 + hh * 16 + 8);
    #pragma unroll
    for (int j = 0; j < 8; ++j) { qa[c][j] = lo[j]; qa[c][j + 8] = hi[j]; }
  }

  v8f o[8];
  #pragma unroll
  for (int nt = 0; nt < 8; ++nt) o[nt] = (v8f){};
  float mrow[8], lrow[8];
  #pragma unroll
  for (int i = 0; i < 8; ++i) { mrow[i] = -3.0e38f; lrow[i] = 0.0f; }

  const _Float16* Kbat = Kb + (size_t)b * S_ * H_;
  const _Float16* Vbat = Vt + (size_t)b * H_ * S_;

  // block-uniform causal bound: last diagonal of the block's 64 queries
  const int kend_blk = ((blockIdx.x * 64) % S_) + 63;

  for (int k0 = 0; k0 <= kend_blk; k0 += 32) {
    // ---- cooperative K/V tile staging: 512 + 512 b128 chunks, 4+4/thread --
    #pragma unroll
    for (int j = 0; j < 4; ++j) {
      int cid = t + j * 128;
      int key = cid >> 4, seg = cid & 15;            // K: 32 rows x 16 chunks
      const _Float16* gk = Kbat + (size_t)(k0 + key) * H_ + seg * 8;
      _Float16* lk = &Klds[key][seg * 8];
      int hv = cid >> 2, sg = cid & 3;               // V: 128 rows x 4 chunks
      const _Float16* gv = Vbat + (size_t)hv * S_ + k0 + sg * 8;
      _Float16* lv = &Vlds[hv][sg * 8];
#ifdef USE_ASYNC_LDS
      async_cp16(gk, lk);
      async_cp16(gv, lv);
#else
      *(v8h*)lk = *(const v8h*)gk;
      *(v8h*)lv = *(const v8h*)gv;
#endif
    }
#ifdef USE_ASYNC_LDS
    asm volatile("s_wait_asynccnt 0" ::: "memory");
#endif
    __syncthreads();

    // ---- scores: two 16x16 tiles (keys k0..+15, k0+16..+31), K=128 --------
    v8f s0 = {}; v8f s1 = {};
    #pragma unroll
    for (int c = 0; c < 4; ++c) {
      v16h bk0 = *(const v16h*)&Klds[n][c * 32 + hh * 16];
      v16h bk1 = *(const v16h*)&Klds[16 + n][c * 32 + hh * 16];
      s0 = __builtin_amdgcn_wmma_f32_16x16x32_f16(false, qa[c], false, bk0, (short)0, s0, false, false);
      s1 = __builtin_amdgcn_wmma_f32_16x16x32_f16(false, qa[c], false, bk1, (short)0, s1, false, false);
    }

    // ---- online softmax; C layout: row = i + 8*hh, col = n ----------------
    float alpha[8];
    #pragma unroll
    for (int i = 0; i < 8; ++i) {
      int qrow_s = qloc + i + 8 * hh;
      float v0 = s0[i];
      float v1 = s1[i];
      // keras mask: keep iff (key <= query) && (score != 0)
      if ((k0 + n)      > qrow_s || v0 == 0.0f) v0 = NEG_INF_;
      if ((k0 + 16 + n) > qrow_s || v1 == 0.0f) v1 = NEG_INF_;
      float mx = fmaxf(v0, v1);
      #pragma unroll
      for (int off = 8; off >= 1; off >>= 1)
        mx = fmaxf(mx, __shfl_xor(mx, off, 32));
      float mnew = fmaxf(mrow[i], mx);
      float p0 = __expf(v0 - mnew);
      float p1 = __expf(v1 - mnew);
      float rs = p0 + p1;
      #pragma unroll
      for (int off = 8; off >= 1; off >>= 1)
        rs += __shfl_xor(rs, off, 32);
      float al = __expf(mrow[i] - mnew);
      lrow[i] = lrow[i] * al + rs;
      mrow[i] = mnew;
      alpha[i] = al;
      Pst[wave][i + 8 * hh][n]      = (_Float16)p0;
      Pst[wave][i + 8 * hh][16 + n] = (_Float16)p1;
    }
    #pragma unroll
    for (int nt = 0; nt < 8; ++nt) {
      #pragma unroll
      for (int i = 0; i < 8; ++i) o[nt][i] *= alpha[i];
    }

    // same-wave cross-lane LDS store -> transposed load fence
    asm volatile("s_wait_dscnt 0" ::: "memory");

    // re-read P as A-operand (16x32)
    v16h pa;
    {
      v8h lo = *(const v8h*)&Pst[wave][n][hh * 16];
      v8h hi = *(const v8h*)&Pst[wave][n][hh * 16 + 8];
      #pragma unroll
      for (int j = 0; j < 8; ++j) { pa[j] = lo[j]; pa[j + 8] = hi[j]; }
    }

    // ---- O += P @ V over 8 output column tiles ----------------------------
    #pragma unroll
    for (int nt = 0; nt < 8; ++nt) {
      v16h bv = *(const v16h*)&Vlds[nt * 16 + n][hh * 16];
      o[nt] = __builtin_amdgcn_wmma_f32_16x16x32_f16(false, pa, false, bv, (short)0, o[nt], false, false);
    }

    __syncthreads();   // all waves done reading K/V tile before next overwrite
  }

  // epilogue: normalize and store fp32 [B,S,H]
  #pragma unroll
  for (int i = 0; i < 8; ++i) {
    float inv = 1.0f / lrow[i];
    int row = growq + i + 8 * hh;
    #pragma unroll
    for (int nt = 0; nt < 8; ++nt)
      out[(size_t)row * H_ + nt * 16 + n] = o[nt][i] * inv;
  }
}

// ---------------------------------------------------------------------------
extern "C" void kernel_launch(void* const* d_in, const int* in_sizes, int n_in,
                              void* d_out, int out_size, void* d_ws, size_t ws_size,
                              hipStream_t stream) {
  (void)in_sizes; (void)n_in; (void)out_size; (void)ws_size;
  const float* x  = (const float*)d_in[0];
  const float* Wq = (const float*)d_in[1];
  const float* Wk = (const float*)d_in[2];
  const float* Wv = (const float*)d_in[3];
  float* out = (float*)d_out;

  _Float16* Wt = (_Float16*)d_ws;          // 3*H*D f16
  _Float16* Qb = Wt + WT_E;                // B*S*H f16 (scaled)
  _Float16* Kb = Qb + QK_E;                // B*S*H f16
  _Float16* Vt = Kb + QK_E;                // B*H*S f16 (transposed)

  hipLaunchKernelGGL(wt_transpose_kernel, dim3((3 * D_ * H_ + 255) / 256),
                     dim3(256), 0, stream, Wq, Wk, Wv, Wt);
  hipLaunchKernelGGL(qkv_proj_kernel, dim3(B_ * S_ / 16),
                     dim3(256), 0, stream, x, Wt, Qb, Kb, Vt);
  hipLaunchKernelGGL(flash_attn_kernel, dim3(B_ * S_ / 16 / 4),
                     dim3(128), 0, stream, Qb, Kb, Vt, out);
}
